// R8ConvF_58961311039804
// MI455X (gfx1250) — matrix operand order repaired and checked by reading, not compiled, not run on hardware
//
#include <hip/hip_runtime.h>

typedef _Float16 half16 __attribute__((ext_vector_type(16)));
typedef _Float16 half8  __attribute__((ext_vector_type(8)));
typedef float    f32x8  __attribute__((ext_vector_type(8)));
typedef float    f32x4  __attribute__((ext_vector_type(4)));

#define TB 256

__device__ __forceinline__ f32x8 wmma_f16(half16 a, half16 b, f32x8 c) {
    // D = A(16x32 f16) x B(32x16 f16) + C(16x16 f32)
    return __builtin_amdgcn_wmma_f32_16x16x32_f16(false, a, false, b,
                                                  (short)0, c, false, false);
}

__global__ __launch_bounds__(TB)
void skel_conv8_wmma(const float* __restrict__ xg,
                     const float* __restrict__ wg,
                     float* __restrict__ outg)
{
    constexpr int C = 128, H = 112, W = 112, O = 128;
    constexpr int HW = H * W;
    constexpr int CP = 136;  // padded channel pitch in halfs (16B-aligned rows)
    // shift index s = (1+dy)*3 + (1+dx); DPLUS[s] = dir for which s is the +dir
    // tap; the opposite dir (DPLUS[s]+4)%8 uses s as its -dir tap.
    constexpr int DPLUS[9] = {5, 6, 7, 4, -1, 0, 3, 2, 1};

    __shared__ __align__(16) _Float16 xs[3 * 18 * CP];     // 14688 B staging
    __shared__ __align__(16) float    obuf[8][2][16][16];  // 16 KB transpose buf

    const int tid  = threadIdx.x;
    const int lane = tid & 31;
    const int wv   = tid >> 5;   // wave id = o-tile (0..7)
    const int hi   = lane >> 4;  // half-wave
    const int lx   = lane & 15;

    const int bh = blockIdx.x;   // 0..16*112-1
    const int b  = bh / H;
    const int h  = bh - b * H;

    // ---- preload 12 B fragments (weight f32 -> f16), kept in VGPRs ----
    // reduction index r = k*128 + c ; chunk j covers r in [32j, 32j+32)
    // B layout: lane holds column N = lx ; element e holds K = e + 16*hi
    half16 Bf[12];
    {
        const int o = wv * 16 + lx;
        #pragma unroll
        for (int j = 0; j < 12; ++j) {
            const int kidx  = j >> 2;
            const int cbase = (j & 3) * 32 + 16 * hi;
            half16 f;
            #pragma unroll
            for (int e = 0; e < 16; ++e)
                f[e] = (_Float16)wg[o * 384 + (cbase + e) * 3 + kidx];
            Bf[j] = f;
        }
    }

    for (int wt = 0; wt < 7; ++wt) {
        const int w0 = wt * 16;

        __syncthreads();
        // ---- stage x 3x18x128 neighborhood (zero-padded) as f16 in LDS ----
        for (int idx = tid; idx < 3 * 18 * C; idx += TB) {
            const int wx = idx % 18;       // fastest: coalesced global w
            const int r  = idx / 18;
            const int c  = r & (C - 1);
            const int sy = r >> 7;
            const int gh = h + sy - 1;
            const int gw = w0 + wx - 1;
            float v = 0.0f;
            if ((unsigned)gh < (unsigned)H && (unsigned)gw < (unsigned)W)
                v = xg[b * (C * HW) + c * HW + gh * W + gw];
            xs[(sy * 18 + wx) * CP + c] = (_Float16)v;
        }
        __syncthreads();

        // ---- compute: 8 dirs x 12 chunks of K=32 ----
        // A layout: lane holds row M = lx ; elem e -> K=16*hi+e
        f32x8 acc[8] = {};

        #pragma unroll
        for (int cc = 0; cc < 4; ++cc) {
            const int cbase = cc * 32 + 16 * hi;

            // center fragment: k=1 tap for all 8 directions
            {
                const _Float16* p = &xs[(1 * 18 + lx + 1) * CP + cbase];
                half8 alo = *(const half8*)p;         // ds_load_b128
                half8 ahi = *(const half8*)(p + 8);   // ds_load_b128
                const half16 Ac = __builtin_shufflevector(alo, ahi,
                        0, 1, 2, 3, 4, 5, 6, 7, 8, 9, 10, 11, 12, 13, 14, 15);
                #pragma unroll
                for (int d = 0; d < 8; ++d)
                    acc[d] = wmma_f16(Ac, Bf[4 + cc], acc[d]);
            }

            // 8 neighbor fragments: each serves 2 opposite directions
            #pragma unroll
            for (int si = 0; si < 8; ++si) {
                const int s  = (si < 4) ? si : si + 1;   // skip center (4)
                const int sy = s / 3, sx = s % 3;
                const _Float16* p = &xs[(sy * 18 + lx + sx) * CP + cbase];
                half8 alo = *(const half8*)p;
                half8 ahi = *(const half8*)(p + 8);
                const half16 As = __builtin_shufflevector(alo, ahi,
                        0, 1, 2, 3, 4, 5, 6, 7, 8, 9, 10, 11, 12, 13, 14, 15);
                const int dp2 = DPLUS[s];
                const int dm2 = (dp2 + 4) & 7;
                acc[dp2] = wmma_f16(As, Bf[8 + cc], acc[dp2]);  // k=2 tap
                acc[dm2] = wmma_f16(As, Bf[0 + cc], acc[dm2]);  // k=0 tap
            }
        }

        // ---- store: transpose 2 dirs at a time through LDS, b128 stores ----
        #pragma unroll 1
        for (int dp = 0; dp < 4; ++dp) {
            #pragma unroll
            for (int q = 0; q < 2; ++q) {
                // D layout: lane -> o_local = lx ; elem v -> w_local = v + 8*hi
                const f32x8 a = acc[dp * 2 + q];
                f32x4 lo = __builtin_shufflevector(a, a, 0, 1, 2, 3);
                f32x4 h4 = __builtin_shufflevector(a, a, 4, 5, 6, 7);
                float* dst = &obuf[wv][q][lx][hi * 8];
                *(f32x4*)dst       = lo;   // ds_store_b128
                *(f32x4*)(dst + 4) = h4;   // ds_store_b128
            }
            __syncthreads();
            // 256 rows (wave x dir x o_local) of 16 floats; 4 lanes per row
            // -> each global_store_b128 group covers contiguous 64B per row
            #pragma unroll
            for (int u = 0; u < 4; ++u) {
                const int unit = tid + u * TB;
                const int row  = unit >> 2;
                const int ch   = unit & 3;
                const int rwv  = row >> 5;
                const int q2   = (row >> 4) & 1;
                const int ol   = row & 15;
                const int d    = dp * 2 + q2;
                const int og   = rwv * 16 + ol;
                const f32x4 v  = *(const f32x4*)&obuf[rwv][q2][ol][ch * 4];
                float* gp = outg + ((size_t)b * (O * 8 * HW)
                                    + (size_t)og * (8 * HW)
                                    + (size_t)d * HW + h * W + w0 + ch * 4);
                *(f32x4*)gp = v;           // global_store_b128
            }
            __syncthreads();
        }
    }
}

extern "C" void kernel_launch(void* const* d_in, const int* in_sizes, int n_in,
                              void* d_out, int out_size, void* d_ws, size_t ws_size,
                              hipStream_t stream)
{
    (void)in_sizes; (void)n_in; (void)out_size; (void)d_ws; (void)ws_size;
    const float* x = (const float*)d_in[0];
    const float* w = (const float*)d_in[1];
    float*     out = (float*)d_out;
    skel_conv8_wmma<<<dim3(16 * 112), dim3(TB), 0, stream>>>(x, w, out);
}
